// GemNetT_53644141527152
// MI455X (gfx1250) — matrix operand order repaired and checked by reading, not compiled, not run on hardware
//
#include <hip/hip_runtime.h>
#include <hip/hip_bf16.h>

typedef __bf16 bf16;
typedef __attribute__((ext_vector_type(16))) __bf16 v16bf;
typedef __attribute__((ext_vector_type(8)))  __bf16 v8bf;
typedef __attribute__((ext_vector_type(8)))  float  v8f;

#define INV_SQRT2F 0.70710678118654752440f
#define LDS_STRIDE 40   // bf16 elements per LDS row (32 data + 8 pad), 80B => 16B-aligned chunks

__device__ __forceinline__ float siluf(float v) { return v / (1.0f + __expf(-v)); }

// ---------------------------------------------------------------------------
// Elementwise / utility kernels
// ---------------------------------------------------------------------------
__global__ void zero_kernel(float* __restrict__ p, long n) {
  long i = (long)blockIdx.x * 256 + threadIdx.x;
  if (i < n) p[i] = 0.0f;
}

__global__ void combine_kernel(float* __restrict__ a, const float* __restrict__ b, long n) {
  long i = (long)blockIdx.x * 256 + threadIdx.x;
  if (i < n) a[i] = (a[i] + b[i]) * INV_SQRT2F;
}

__global__ void add_kernel(float* __restrict__ a, const float* __restrict__ b, long n) {
  long i = (long)blockIdx.x * 256 + threadIdx.x;
  if (i < n) a[i] += b[i];
}

// rbf[e,r] = envelope(d/6) * exp(-0.5/(1/127)^2 * (d/6 - r/127)^2)
__global__ void rbf_kernel(const float* __restrict__ D_st, float* __restrict__ rbf, int E) {
  long idx = (long)blockIdx.x * 256 + threadIdx.x;
  if (idx >= (long)E * 128) return;
  int e = (int)(idx >> 7);
  int r = (int)(idx & 127);
  float ds  = D_st[e] * (1.0f / 6.0f);
  float off = (float)r * (1.0f / 127.0f);
  float d   = ds - off;
  float g   = __expf(-8064.5f * d * d);   // -0.5*127^2
  float d2 = ds * ds, d4 = d2 * d2, d5 = d4 * ds, d6 = d5 * ds, d7 = d6 * ds;
  float env = (ds < 1.0f) ? (1.0f - 21.0f * d5 + 35.0f * d6 - 15.0f * d7) : 0.0f;
  rbf[idx] = env * g;
}

// sph[t, l] = sqrt((2l+1)/4pi) * P_l(cos)
__global__ void sph_kernel(const float* __restrict__ cosphi, float* __restrict__ sph, int T) {
  int t = blockIdx.x * 256 + threadIdx.x;
  if (t >= T) return;
  float x = cosphi[t];
  float P[7];
  P[0] = 1.0f; P[1] = x;
#pragma unroll
  for (int l = 1; l < 6; l++)
    P[l + 1] = ((float)(2 * l + 1) * x * P[l] - (float)l * P[l - 1]) / (float)(l + 1);
  const float cs[7] = {0.2820947917738781f, 0.4886025119029199f, 0.6307831305050401f,
                       0.7463526651802308f, 0.8462843753216345f, 0.9356025796273889f,
                       1.0171072362820548f};
  float* o = sph + (size_t)t * 7;
#pragma unroll
  for (int s = 0; s < 7; s++) o[s] = cs[s] * P[s];
}

__global__ void gather_h_kernel(const float* __restrict__ table, const int* __restrict__ z,
                                float* __restrict__ h, int A) {
  long idx = (long)blockIdx.x * 256 + threadIdx.x;
  if (idx >= (long)A * 128) return;
  int a = (int)(idx >> 7);
  int c = (int)(idx & 127);
  h[idx] = table[(size_t)z[a] * 128 + c];
}

// dst[idx_t[e], c] += m[e,c] * coef[e,c]
__global__ void edge_scatter_atom(const float* __restrict__ m, const float* __restrict__ coef,
                                  const int* __restrict__ idxt, float* __restrict__ dst, int E) {
  long idx = (long)blockIdx.x * 256 + threadIdx.x;
  if (idx >= (long)E * 128) return;
  int e = (int)(idx >> 7);
  int c = (int)(idx & 127);
  atomicAdd(&dst[(size_t)idxt[e] * 128 + c], m[idx] * coef[idx]);
}

// sum_t[id3_ca[tr], s, t] += sph[tr, s] * xdown[id3_ba[tr], t]
__global__ void trip_scatter(const float* __restrict__ sph, const float* __restrict__ xdown,
                             const int* __restrict__ id3_ba, const int* __restrict__ id3_ca,
                             float* __restrict__ sum_t, int T) {
  int trip = blockIdx.x * 4 + (threadIdx.x >> 6);
  int t = threadIdx.x & 63;
  if (trip >= T) return;
  int eba = id3_ba[trip];
  int eca = id3_ca[trip];
  float v = xdown[(size_t)eba * 64 + t];
  const float* sp = sph + (size_t)trip * 7;
  float* dst = sum_t + (size_t)eca * 448 + t;
#pragma unroll
  for (int s = 0; s < 7; s++) atomicAdd(dst + s * 64, sp[s] * v);
}

// inter[e, c*64+t] = sum_s rbfW1[e, s*16+c] * sum_t[e, s*64+t]     (stored bf16)
__global__ void inter_kernel(const float* __restrict__ rbfW1, const float* __restrict__ st,
                             bf16* __restrict__ inter, long n) {
  long idx = (long)blockIdx.x * 256 + threadIdx.x;
  if (idx >= n) return;
  long e = idx >> 10;
  int ct = (int)(idx & 1023);
  int c = ct >> 6, t = ct & 63;
  const float* rw = rbfW1 + e * 112 + c;
  const float* sp = st + e * 448 + t;
  float acc = 0.0f;
#pragma unroll
  for (int s = 0; s < 7; s++) acc = fmaf(rw[s * 16], sp[s * 64], acc);
  inter[idx] = (bf16)acc;
}

// Weight repack: f32 [K,N] (row stride ldw) -> bf16 [N,K]
__global__ void repack_w(const float* __restrict__ W, int ldw, bf16* __restrict__ Wt,
                         int K, int N) {
  long idx = (long)blockIdx.x * 256 + threadIdx.x;
  if (idx >= (long)N * K) return;
  int n = (int)(idx / K);
  int k = (int)(idx % K);
  Wt[idx] = (bf16)W[(size_t)k * ldw + n];
}

// W_bil [64][16][64] (t,c,b) -> bf16 Wt[b][c*64+t]  (i.e. [N=64][K=1024])
__global__ void repack_wbil(const float* __restrict__ wbil, bf16* __restrict__ wt) {
  int idx = blockIdx.x * 256 + threadIdx.x;   // 65536
  int b = idx >> 10;
  int ct = idx & 1023;
  int c = ct >> 6, t = ct & 63;
  wt[idx] = (bf16)wbil[((size_t)t * 16 + c) * 64 + b];
}

__global__ void energy_kernel(const float* __restrict__ x, const float* __restrict__ w,
                              float* __restrict__ Eacc, int A) {
  int a = blockIdx.x * 256 + threadIdx.x;
  if (a >= A) return;
  const float* xr = x + (size_t)a * 128;
  float acc = 0.0f;
#pragma unroll 16
  for (int c = 0; c < 128; c++) acc = fmaf(xr[c], w[c], acc);
  Eacc[a] += acc;
}

__global__ void reduce_sum(const float* __restrict__ v, int n, float* __restrict__ out) {
  __shared__ float sm[256];
  int i = blockIdx.x * 256 + threadIdx.x;
  sm[threadIdx.x] = (i < n) ? v[i] : 0.0f;
  __syncthreads();
  for (int s = 128; s > 0; s >>= 1) {
    if ((int)threadIdx.x < s) sm[threadIdx.x] += sm[threadIdx.x + s];
    __syncthreads();
  }
  if (threadIdx.x == 0) atomicAdd(out, sm[0]);
}

// ---------------------------------------------------------------------------
// Generic WMMA bf16 GEMM:  D[M,N] = epi( A[M,K] @ W[K,N] )
//   Weight is pre-transposed/converted: Wt = bf16 [N][K].
//   EPI: 0 none | 1 silu | 2 silu*aux | 3 (aux + silu)*INV_SQRT2
//   SRC: 0 A=f32 | 1 A=concat(h[idx_s],h[idx_t],third) | 2 A=bf16
//   NT : number of 16-wide N tiles per block (1 or 4)
// Block: 128 threads = 4 waves; tile 64 x (NT*16); K stepped by 32.
// NOTE: all K used here are multiples of 16, and concat segment boundaries
// (128/256) are multiples of 16, so every 16-element staging chunk is either
// fully in-bounds or fully out -> chunk-level guards only (vectorizable).
// ---------------------------------------------------------------------------
template <int EPI, int SRC, int NT>
__global__ __launch_bounds__(128)
void gemm_wmma(const void* __restrict__ Aptr, int lda,
               const bf16* __restrict__ Wt,
               float* __restrict__ D, int ldd,
               const float* __restrict__ aux, int ldaux,
               const float* __restrict__ hsrc,
               const int* __restrict__ gidx_s, const int* __restrict__ gidx_t,
               int M, int N, int K) {
  __shared__ bf16 As[64 * LDS_STRIDE];
  __shared__ bf16 Bs[64 * LDS_STRIDE];
  const int tid  = threadIdx.x;
  const int wave = tid >> 5;
  const int lane = tid & 31;
  const int half = lane >> 4;
  const int lrow = lane & 15;
  const int blockM = (int)blockIdx.y * 64;
  const int blockN = (int)blockIdx.x * (NT * 16);

  v8f acc[NT];
#pragma unroll
  for (int j = 0; j < NT; j++)
#pragma unroll
    for (int e = 0; e < 8; e++) acc[j][e] = 0.0f;

  // staging coordinates: each thread fills one 16-element K-chunk of one A row / one B row
  const int srow = tid >> 1;
  const int skL  = (tid & 1) * 16;
  const int agm  = blockM + srow;
  const bool mok = agm < M;
  const float* arow = nullptr;
  const bf16*  arowb = nullptr;
  const float* hs = nullptr;
  const float* ht = nullptr;
  const float* trow = nullptr;
  if constexpr (SRC == 0) {
    if (mok) arow = (const float*)Aptr + (size_t)agm * lda;
  } else if constexpr (SRC == 2) {
    if (mok) arowb = (const bf16*)Aptr + (size_t)agm * lda;
  } else {
    if (mok) {
      hs   = hsrc + (size_t)gidx_s[agm] * 128;
      ht   = hsrc + (size_t)gidx_t[agm] * 128;
      trow = (const float*)Aptr + (size_t)agm * lda;
    }
  }
  const int bgn  = blockN + srow;
  const bool nok = bgn < N;
  const bf16* wrow = nok ? (Wt + (size_t)bgn * K) : nullptr;

  v8bf zv;
#pragma unroll
  for (int e = 0; e < 8; e++) zv[e] = (bf16)0.0f;

  const int KT = (K + 31) >> 5;
  for (int kt = 0; kt < KT; kt++) {
    const int k0 = kt << 5;
    const int kc = k0 + skL;          // chunk start; fully in-bounds iff kc < K
    // --- stage A chunk (16 bf16 = 32B, two 16B LDS stores) ---
    {
      v8bf* dst = (v8bf*)&As[srow * LDS_STRIDE + skL];
      if (mok && kc < K) {
        if constexpr (SRC == 2) {
          const v8bf* s = (const v8bf*)(arowb + kc);
          dst[0] = s[0];
          dst[1] = s[1];
        } else {
          const float* src;
          if constexpr (SRC == 0) src = arow + kc;
          else src = (kc < 128) ? (hs + kc) : ((kc < 256) ? (ht + (kc - 128)) : (trow + (kc - 256)));
          v8bf lo, hi;
#pragma unroll
          for (int i = 0; i < 8; i++) { lo[i] = (bf16)src[i]; hi[i] = (bf16)src[8 + i]; }
          dst[0] = lo;
          dst[1] = hi;
        }
      } else {
        dst[0] = zv;
        dst[1] = zv;
      }
    }
    // --- stage B chunk from pre-transposed bf16 weight [N][K] ---
    {
      v8bf* dst = (v8bf*)&Bs[srow * LDS_STRIDE + skL];
      if (nok && kc < K) {
        const v8bf* s = (const v8bf*)(wrow + kc);
        dst[0] = s[0];
        dst[1] = s[1];
      } else {
        dst[0] = zv;
        dst[1] = zv;
      }
    }
    __syncthreads();

    // --- A fragment (ISA 16-bit A 16x32): lane<16 K=[0..7,16..23], lane>=16 K=[8..15,24..31]
    const bf16* ap = &As[(wave * 16 + lrow) * LDS_STRIDE + 8 * half];
    v8bf alo = *(const v8bf*)(ap);
    v8bf ahi = *(const v8bf*)(ap + 16);
    v16bf afrag;
#pragma unroll
    for (int e = 0; e < 8; e++) { afrag[e] = alo[e]; afrag[e + 8] = ahi[e]; }

#pragma unroll
    for (int j = 0; j < NT; j++) {
      // --- B fragment (32x16): lanes 0-15 K=[0..7,16..23], lanes 16-31 K=[8..15,24..31]
      const bf16* bp = &Bs[(j * 16 + lrow) * LDS_STRIDE + 8 * half];
      v8bf b0 = *(const v8bf*)(bp);
      v8bf b1 = *(const v8bf*)(bp + 16);
      v16bf bfrag;
#pragma unroll
      for (int e = 0; e < 8; e++) { bfrag[e] = b0[e]; bfrag[e + 8] = b1[e]; }
      acc[j] = __builtin_amdgcn_wmma_f32_16x16x32_bf16(false, afrag, false, bfrag,
                                                       (short)0, acc[j], false, false);
    }
    __syncthreads();
  }

  // --- epilogue: C/D layout lane: N = lane%16, element jj: M = jj + 8*(lane/16)
  const int gmBase = blockM + wave * 16 + 8 * half;
#pragma unroll
  for (int j = 0; j < NT; j++) {
    const int gn = blockN + j * 16 + lrow;
    if (gn >= N) continue;
#pragma unroll
    for (int jj = 0; jj < 8; jj++) {
      const int gm = gmBase + jj;
      if (gm >= M) continue;
      float v = acc[j][jj];
      if constexpr (EPI == 1) v = siluf(v);
      else if constexpr (EPI == 2) v = siluf(v) * aux[(size_t)gm * ldaux + gn];
      else if constexpr (EPI == 3) v = (aux[(size_t)gm * ldaux + gn] + siluf(v)) * INV_SQRT2F;
      D[(size_t)gm * ldd + gn] = v;
    }
  }
}

// ---------------------------------------------------------------------------
// Host orchestration
// ---------------------------------------------------------------------------
#define LAUNCH_GEMM(EPI, SRC, NT, Aq, lda_, W_, ldw_, D_, ldd_, aux_, ldaux_, hs_, is_, it_, M_, N_, K_) \
  do {                                                                                                   \
    repack_w<<<(unsigned)((((long)(N_) * (K_)) + 255) / 256), 256, 0, stream>>>(                         \
        (W_), (ldw_), Wt, (K_), (N_));                                                                   \
    gemm_wmma<EPI, SRC, NT><<<dim3((unsigned)(((N_) + ((NT)*16) - 1) / ((NT)*16)),                       \
                                   (unsigned)(((M_) + 63) / 64)),                                        \
                              dim3(128), 0, stream>>>(                                                   \
        (const void*)(Aq), (lda_), Wt, (D_), (ldd_), (aux_), (ldaux_), (hs_), (is_), (it_),              \
        (M_), (N_), (K_));                                                                               \
  } while (0)

extern "C" void kernel_launch(void* const* d_in, const int* in_sizes, int n_in,
                              void* d_out, int out_size, void* d_ws, size_t ws_size,
                              hipStream_t stream) {
  (void)n_in; (void)out_size; (void)ws_size;
  const int E = in_sizes[0];
  const int T = in_sizes[1];
  const int A = in_sizes[2];

  const float* D_st       = (const float*)d_in[0];
  const float* cosphi     = (const float*)d_in[1];
  const int*   zatom      = (const int*)d_in[2];
  const int*   idx_s      = (const int*)d_in[3];
  const int*   idx_t      = (const int*)d_in[4];
  const int*   id3_ba     = (const int*)d_in[5];
  const int*   id3_ca     = (const int*)d_in[6];
  const float* atom_table = (const float*)d_in[7];
  const float* W_edge     = (const float*)d_in[8];
  const float* W_rbf3     = (const float*)d_in[9];
  const float* W_cbf3     = (const float*)d_in[10];
  const float* W_rbf_h    = (const float*)d_in[11];
  const float* W_rbf_out  = (const float*)d_in[12];
  const float* W_ca       = (const float*)d_in[13];
  const float* W_ba       = (const float*)d_in[14];
  const float* W_rbf_trip = (const float*)d_in[15];
  const float* W_down     = (const float*)d_in[16];
  const float* W_bil      = (const float*)d_in[17];
  const float* W_up       = (const float*)d_in[18];
  const float* Wres_pre   = (const float*)d_in[19];
  const float* Wres_post  = (const float*)d_in[20];
  const float* W_ah_rbf   = (const float*)d_in[21];
  const float* W_ah0      = (const float*)d_in[22];
  const float* Wres_atom  = (const float*)d_in[23];
  const float* W_concat   = (const float*)d_in[24];
  const float* Wres_cat   = (const float*)d_in[25];
  const float* W_out_rbf  = (const float*)d_in[26];
  const float* W_out0     = (const float*)d_in[27];
  const float* Wres_out   = (const float*)d_in[28];
  const float* W_energy   = (const float*)d_in[29];
  float*       Eout       = (float*)d_out;

  // ---- workspace carve-out ----
  char* wsb = (char*)d_ws;
  size_t off = 0;
  auto alloc = [&](size_t bytes) -> char* {
    char* p = wsb + off;
    off = (off + bytes + 255) & ~(size_t)255;
    return p;
  };
  float* rbf    = (float*)alloc((size_t)E * 128 * 4);  // reused as coefE after phase 0
  float* rbf3B  = (float*)alloc((size_t)E * 16 * 4);
  float* rbfhB  = (float*)alloc((size_t)E * 16 * 4);
  float* rbfoB  = (float*)alloc((size_t)E * 16 * 4);
  float* rbfW1  = (float*)alloc((size_t)E * 112 * 4);
  float* sphB   = (float*)alloc((size_t)T * 7 * 4);
  float* hB     = (float*)alloc((size_t)A * 128 * 4);
  float* mB     = (float*)alloc((size_t)E * 128 * 4);
  float* e1     = (float*)alloc((size_t)E * 128 * 4);
  float* e2     = (float*)alloc((size_t)E * 128 * 4);
  float* e3     = (float*)alloc((size_t)E * 128 * 4);
  float* xdown  = (float*)alloc((size_t)E * 64 * 4);
  float* sumT   = (float*)alloc((size_t)E * 448 * 4);
  bf16*  interB = (bf16*) alloc((size_t)E * 1024 * 2);
  float* xa     = (float*)alloc((size_t)A * 128 * 4);
  float* at1    = (float*)alloc((size_t)A * 128 * 4);
  float* at2    = (float*)alloc((size_t)A * 128 * 4);
  float* Eacc   = (float*)alloc((size_t)A * 4);
  bf16*  Wt     = (bf16*) alloc((size_t)65536 * 2);   // transposed bf16 weight scratch
  float* coefE  = rbf;  // alias; rbf no longer needed after phase 0

  auto zero = [&](float* p, long n) {
    zero_kernel<<<(unsigned)((n + 255) / 256), 256, 0, stream>>>(p, n);
  };

  // residual stack: for each layer: x = (x + silu(silu(x@W0)@W1)) * inv_sqrt2
  auto residual = [&](const float* Wres, int nl, float* x, float* tmp, int M) {
    for (int l = 0; l < nl; l++) {
      const float* W0 = Wres + (size_t)l * 2 * 128 * 128;
      const float* W1 = W0 + 128 * 128;
      LAUNCH_GEMM(1, 0, 4, x, 128, W0, 128, tmp, 128, nullptr, 0, nullptr, nullptr, nullptr, M, 128, 128);
      LAUNCH_GEMM(3, 0, 4, tmp, 128, W1, 128, x, 128, x, 128, nullptr, nullptr, nullptr, M, 128, 128);
    }
  };

  auto out_block = [&](int i, const float* mcur) {
    // coef = rbf_out @ W_out_rbf[i]    [E,128], K=16
    LAUNCH_GEMM(0, 0, 4, rbfoB, 16, W_out_rbf + (size_t)i * 16 * 128, 128, coefE, 128,
                nullptr, 0, nullptr, nullptr, nullptr, E, 128, 16);
    zero(xa, (long)A * 128);
    edge_scatter_atom<<<(unsigned)(((long)E * 128 + 255) / 256), 256, 0, stream>>>(
        mcur, coefE, idx_t, xa, E);
    LAUNCH_GEMM(1, 0, 4, xa, 128, W_out0 + (size_t)i * 128 * 128, 128, at1, 128,
                nullptr, 0, nullptr, nullptr, nullptr, A, 128, 128);
    residual(Wres_out + (size_t)i * 6 * 128 * 128, 3, at1, at2, A);
    energy_kernel<<<(unsigned)((A + 255) / 256), 256, 0, stream>>>(
        at1, W_energy + (size_t)i * 128, Eacc, A);
  };

  // ================= Phase 0: bases + initial embeddings =================
  zero(Eout, 1);
  zero(Eacc, A);
  rbf_kernel<<<(unsigned)(((long)E * 128 + 255) / 256), 256, 0, stream>>>(D_st, rbf, E);
  sph_kernel<<<(unsigned)((T + 255) / 256), 256, 0, stream>>>(cosphi, sphB, T);
  gather_h_kernel<<<(unsigned)(((long)A * 128 + 255) / 256), 256, 0, stream>>>(
      atom_table, zatom, hB, A);

  LAUNCH_GEMM(0, 0, 1, rbf, 128, W_rbf3, 16, rbf3B, 16, nullptr, 0, nullptr, nullptr, nullptr, E, 16, 128);
  LAUNCH_GEMM(0, 0, 1, rbf, 128, W_rbf_h, 16, rbfhB, 16, nullptr, 0, nullptr, nullptr, nullptr, E, 16, 128);
  LAUNCH_GEMM(0, 0, 1, rbf, 128, W_rbf_out, 16, rbfoB, 16, nullptr, 0, nullptr, nullptr, nullptr, E, 16, 128);
  for (int s = 0; s < 7; s++)
    LAUNCH_GEMM(0, 0, 1, rbf, 128, W_cbf3 + (size_t)s * 128 * 16, 16, rbfW1 + s * 16, 112,
                nullptr, 0, nullptr, nullptr, nullptr, E, 16, 128);

  // m = silu(concat(h[idx_s], h[idx_t], rbf) @ W_edge)
  LAUNCH_GEMM(1, 1, 4, rbf, 128, W_edge, 128, mB, 128, nullptr, 0, hB, idx_s, idx_t, E, 128, 384);

  out_block(0, mB);

  // ================= NB interaction blocks =================
  for (int i = 0; i < 3; i++) {
    // x_ca = silu(m @ W_ca)
    LAUNCH_GEMM(1, 0, 4, mB, 128, W_ca + (size_t)i * 128 * 128, 128, e1, 128,
                nullptr, 0, nullptr, nullptr, nullptr, E, 128, 128);
    // r3t = rbf3 @ W_rbf_trip  (K=16)
    LAUNCH_GEMM(0, 0, 4, rbf3B, 16, W_rbf_trip + (size_t)i * 16 * 128, 128, coefE, 128,
                nullptr, 0, nullptr, nullptr, nullptr, E, 128, 16);
    // x_ba = silu(m @ W_ba) * r3t
    LAUNCH_GEMM(2, 0, 4, mB, 128, W_ba + (size_t)i * 128 * 128, 128, e2, 128,
                coefE, 128, nullptr, nullptr, nullptr, E, 128, 128);
    // xdown = silu(x_ba @ W_down)  [E,64]
    LAUNCH_GEMM(1, 0, 4, e2, 128, W_down + (size_t)i * 128 * 64, 64, xdown, 64,
                nullptr, 0, nullptr, nullptr, nullptr, E, 64, 128);
    // triplet scatter: sum_t[E,7,64]
    zero(sumT, (long)E * 448);
    trip_scatter<<<(unsigned)((T + 3) / 4), 256, 0, stream>>>(sphB, xdown, id3_ba, id3_ca, sumT, T);
    // inter[E,16,64] (bf16)
    inter_kernel<<<(unsigned)(((long)E * 1024 + 255) / 256), 256, 0, stream>>>(
        rbfW1, sumT, interB, (long)E * 1024);
    // x3pre = inter @ W_bil'   (K=1024, N=64); W_bil repacked+transposed to bf16 [64][1024]
    repack_wbil<<<256, 256, 0, stream>>>(W_bil + (size_t)i * 64 * 16 * 64, Wt);
    gemm_wmma<0, 2, 4><<<dim3(1, (unsigned)((E + 63) / 64)), dim3(128), 0, stream>>>(
        (const void*)interB, 1024, Wt, e2, 64, nullptr, 0, nullptr, nullptr, nullptr, E, 64, 1024);
    // x = (x_ca + silu(x3pre @ W_up)) * inv_sqrt2
    LAUNCH_GEMM(3, 0, 4, e2, 64, W_up + (size_t)i * 64 * 128, 128, e1, 128,
                e1, 128, nullptr, nullptr, nullptr, E, 128, 64);
    // residual pre (1 layer)
    residual(Wres_pre + (size_t)i * 2 * 128 * 128, 1, e1, e3, E);
    // m = (m + x) * inv_sqrt2
    combine_kernel<<<(unsigned)(((long)E * 128 + 255) / 256), 256, 0, stream>>>(mB, e1, (long)E * 128);
    // residual post (2 layers)
    residual(Wres_post + (size_t)i * 4 * 128 * 128, 2, mB, e3, E);

    // ---- atom update ----
    LAUNCH_GEMM(0, 0, 4, rbfhB, 16, W_ah_rbf + (size_t)i * 16 * 128, 128, coefE, 128,
                nullptr, 0, nullptr, nullptr, nullptr, E, 128, 16);
    zero(xa, (long)A * 128);
    edge_scatter_atom<<<(unsigned)(((long)E * 128 + 255) / 256), 256, 0, stream>>>(
        mB, coefE, idx_t, xa, E);
    LAUNCH_GEMM(1, 0, 4, xa, 128, W_ah0 + (size_t)i * 128 * 128, 128, at1, 128,
                nullptr, 0, nullptr, nullptr, nullptr, A, 128, 128);
    residual(Wres_atom + (size_t)i * 6 * 128 * 128, 3, at1, at2, A);
    add_kernel<<<(unsigned)(((long)A * 128 + 255) / 256), 256, 0, stream>>>(hB, at1, (long)A * 128);

    // ---- concat block: m = residual(silu(concat(h_s,h_t,m) @ W_concat)) ----
    LAUNCH_GEMM(1, 1, 4, mB, 128, W_concat + (size_t)i * 384 * 128, 128, e1, 128,
                nullptr, 0, hB, idx_s, idx_t, E, 128, 384);
    {
      const float* W0 = Wres_cat + (size_t)i * 2 * 128 * 128;
      const float* W1 = W0 + 128 * 128;
      LAUNCH_GEMM(1, 0, 4, e1, 128, W0, 128, e3, 128, nullptr, 0, nullptr, nullptr, nullptr, E, 128, 128);
      LAUNCH_GEMM(3, 0, 4, e3, 128, W1, 128, mB, 128, e1, 128, nullptr, nullptr, nullptr, E, 128, 128);
    }

    out_block(i + 1, mB);
  }

  // ================= final reduce: sum per-atom energies =================
  reduce_sum<<<(unsigned)((A + 255) / 256), 256, 0, stream>>>(Eacc, A, Eout);
}
